// Model_24429773980161
// MI455X (gfx1250) — matrix operand order repaired and checked by reading, not compiled, not run on hardware
//
#include <hip/hip_runtime.h>
#include <hip/hip_bf16.h>
#include <math.h>

#define B_      16
#define NNODES  1024
#define NEDGES  256
#define NINC    4096
#define C_      512
#define OUT_ELEMS (B_*NNODES*C_)      // 8388608
#define NEG_SLOPE 0.2f
#define MARGIN    4.2f
#define EPS_SM    1e-16f

typedef __attribute__((ext_vector_type(16))) __bf16        v16bf;
typedef __attribute__((ext_vector_type(8)))  float         v8f;
typedef __attribute__((ext_vector_type(4)))  unsigned int  u32x4;

// ---------------------------------------------------------------- helpers
static __device__ __forceinline__ unsigned short f32_to_bf16(float f) {
    unsigned int u = __float_as_uint(f);
    u += 0x7FFFu + ((u >> 16) & 1u);          // round-to-nearest-even
    return (unsigned short)(u >> 16);
}

// A fragment (16x32 bf16, MxK): lane l -> row m=l&15, K = (l>>4)*8 + {0..7, 16..23}
static __device__ __forceinline__ v16bf load_a_frag(const unsigned short* row_k0, int lane) {
    union { v16bf v; u32x4 q[2]; } f;
    const unsigned short* p = row_k0 + ((lane >> 4) << 3);
    f.q[0] = *(const u32x4*)(p);
    f.q[1] = *(const u32x4*)(p + 16);
    return f.v;
}
// B fragment (32x16 bf16, KxN) from row-major transposed source [N][K]:
// lane l -> col n=l&15, K = (l>>4)*8 + {0..7, 16..23}
static __device__ __forceinline__ v16bf load_b_frag(const unsigned short* row_k0, int lane) {
    union { v16bf v; u32x4 q[2]; } f;
    const unsigned short* p = row_k0 + ((lane >> 4) << 3);
    f.q[0] = *(const u32x4*)(p);
    f.q[1] = *(const u32x4*)(p + 16);
    return f.v;
}

static __device__ __forceinline__ int lower_bound_i(const int* arr, int n, int key) {
    int lo = 0, hi = n;
    while (lo < hi) { int mid = (lo + hi) >> 1; if (arr[mid] < key) lo = mid + 1; else hi = mid; }
    return lo;
}

static __device__ __forceinline__ void atomicMaxFloat(float* addr, float v) {
    if (v >= 0.0f) atomicMax((int*)addr, __float_as_int(v));
    else           atomicMin((unsigned int*)addr, __float_as_uint(v));
}

// ---------------------------------------------------------------- fills / converts
__global__ void fill_u32_kernel(unsigned int* p, size_t n, unsigned int v) {
    size_t i = (size_t)blockIdx.x * blockDim.x + threadIdx.x;
    size_t stride = (size_t)gridDim.x * blockDim.x;
    for (; i < n; i += stride) p[i] = v;
}

__global__ void convert_x_kernel(const float* __restrict__ x, unsigned short* __restrict__ xbf) {
    size_t i = (size_t)blockIdx.x * blockDim.x + threadIdx.x;
    size_t stride = (size_t)gridDim.x * blockDim.x;
    for (; i < (size_t)OUT_ELEMS; i += stride) xbf[i] = f32_to_bf16(x[i]);
}

// W [K][C] f32 -> Wt [C][K] bf16 (row-major transposed for B-fragment loads)
__global__ void convert_wt_kernel(const float* __restrict__ w, unsigned short* __restrict__ wt) {
    int i = blockIdx.x * blockDim.x + threadIdx.x;   // exact 512*512
    int c = i >> 9, k = i & 511;
    wt[(size_t)c * C_ + k] = f32_to_bf16(w[(size_t)k * C_ + c]);
}

// ---------------------------------------------------------------- GEMM 1: xw = x @ W -> x1 [N][B][C] f32
// Register-blocked: each wave computes a 32x64 output block (2 M-tiles x 4 N-tiles),
// 8 WMMAs per k-step fed by 12 b128 loads (vs 1 WMMA / 4 loads unblocked).
__global__ __launch_bounds__(256) void gemm1_kernel(const unsigned short* __restrict__ xbf,
                                                    const unsigned short* __restrict__ wt,
                                                    float* __restrict__ x1) {
    int w = (blockIdx.x << 3) + (threadIdx.x >> 5);   // 4096 waves
    int lane = threadIdx.x & 31;
    int mt = w >> 3;                                  // [0,512): 32-row block
    int nt = w & 7;                                   // [0,8):   64-col block
    int mi = lane & 15;
    const unsigned short* arow0 = xbf + (size_t)(mt * 32 + mi) * C_;
    const unsigned short* arow1 = arow0 + (size_t)16 * C_;
    const unsigned short* brow  = wt  + (size_t)(nt * 64 + mi) * C_;
    v8f acc[8] = {};                                  // acc[i*4+j]: i=m-subtile, j=n-subtile
#pragma unroll
    for (int k0 = 0; k0 < C_; k0 += 32) {
        v16bf a0 = load_a_frag(arow0 + k0, lane);
        v16bf a1 = load_a_frag(arow1 + k0, lane);
        v16bf b0 = load_b_frag(brow + k0, lane);
        v16bf b1 = load_b_frag(brow + (size_t)16 * C_ + k0, lane);
        v16bf b2 = load_b_frag(brow + (size_t)32 * C_ + k0, lane);
        v16bf b3 = load_b_frag(brow + (size_t)48 * C_ + k0, lane);
        acc[0] = __builtin_amdgcn_wmma_f32_16x16x32_bf16(false, a0, false, b0, (short)0, acc[0], false, false);
        acc[1] = __builtin_amdgcn_wmma_f32_16x16x32_bf16(false, a0, false, b1, (short)0, acc[1], false, false);
        acc[2] = __builtin_amdgcn_wmma_f32_16x16x32_bf16(false, a0, false, b2, (short)0, acc[2], false, false);
        acc[3] = __builtin_amdgcn_wmma_f32_16x16x32_bf16(false, a0, false, b3, (short)0, acc[3], false, false);
        acc[4] = __builtin_amdgcn_wmma_f32_16x16x32_bf16(false, a1, false, b0, (short)0, acc[4], false, false);
        acc[5] = __builtin_amdgcn_wmma_f32_16x16x32_bf16(false, a1, false, b1, (short)0, acc[5], false, false);
        acc[6] = __builtin_amdgcn_wmma_f32_16x16x32_bf16(false, a1, false, b2, (short)0, acc[6], false, false);
        acc[7] = __builtin_amdgcn_wmma_f32_16x16x32_bf16(false, a1, false, b3, (short)0, acc[7], false, false);
    }
    int mh = (lane >> 4) * 8;
#pragma unroll
    for (int i = 0; i < 2; i++) {
#pragma unroll
        for (int j = 0; j < 4; j++) {
#pragma unroll
            for (int v = 0; v < 8; v++) {
                int r = mt * 32 + i * 16 + v + mh;    // r = b*1024 + n
                int b = r >> 10, n = r & 1023;
                int c = nt * 64 + j * 16 + mi;
                x1[((size_t)n * B_ + b) * C_ + c] = acc[i * 4 + j][v];
            }
        }
    }
}

// ---------------------------------------------------------------- edge segment sum (+deg, Bn, bf16 copy)
__global__ __launch_bounds__(256) void edge_sums_kernel(const float* __restrict__ x1,
                                                        const int* __restrict__ node_idx,
                                                        const int* __restrict__ edge_idx,
                                                        float* __restrict__ esum,
                                                        unsigned short* __restrict__ esb,
                                                        float* __restrict__ Bn) {
    __shared__ int snode[NINC];
    int m  = blockIdx.x;
    int lo = lower_bound_i(edge_idx, NINC, m);
    int hi = lower_bound_i(edge_idx, NINC, m + 1);
    int cnt = hi - lo;
    for (int i = threadIdx.x; i < cnt; i += blockDim.x) snode[i] = node_idx[lo + i];
    __syncthreads();
    if (threadIdx.x == 0) Bn[m] = cnt > 0 ? 1.0f / fmaxf((float)cnt, 1.0f) : 0.0f;
    int t = threadIdx.x;                              // 2 channels per thread
    for (int b = 0; b < B_; b++) {
        float a0 = 0.0f, a1 = 0.0f;
        for (int i = 0; i < cnt; i++) {
            const float* row = x1 + ((size_t)snode[i] * B_ + b) * C_;
            a0 += row[2 * t]; a1 += row[2 * t + 1];
        }
        size_t o  = ((size_t)m * B_ + b) * C_ + 2 * t;
        esum[o] = a0; esum[o + 1] = a1;
        size_t ob = ((size_t)b * NEDGES + m) * C_ + 2 * t;
        esb[ob] = f32_to_bf16(a0); esb[ob + 1] = f32_to_bf16(a1);
    }
}

// ---------------------------------------------------------------- GEMM 2: ip[b] = G_b G_b^T (256x256 per batch)
// Register-blocked 32x32 (2x2 tiles): 4 WMMAs per k-step from 8 b128 loads.
__global__ __launch_bounds__(256) void gemm2_kernel(const unsigned short* __restrict__ esb,
                                                    float* __restrict__ ip) {
    int w = (blockIdx.x << 3) + (threadIdx.x >> 5);   // 1024 waves
    int lane = threadIdx.x & 31;
    int bb = w >> 6;                                  // batch
    int tt = w & 63;
    int mt = tt >> 3, nt = tt & 7;                    // 8x8 blocks of 32x32
    int mi = lane & 15;
    const unsigned short* base  = esb + (size_t)bb * NEDGES * C_;
    const unsigned short* arow0 = base + (size_t)(mt * 32 + mi) * C_;
    const unsigned short* arow1 = arow0 + (size_t)16 * C_;
    const unsigned short* brow0 = base + (size_t)(nt * 32 + mi) * C_;
    const unsigned short* brow1 = brow0 + (size_t)16 * C_;
    v8f acc[4] = {};
#pragma unroll
    for (int k0 = 0; k0 < C_; k0 += 32) {
        v16bf a0 = load_a_frag(arow0 + k0, lane);
        v16bf a1 = load_a_frag(arow1 + k0, lane);
        v16bf b0 = load_b_frag(brow0 + k0, lane);
        v16bf b1 = load_b_frag(brow1 + k0, lane);
        acc[0] = __builtin_amdgcn_wmma_f32_16x16x32_bf16(false, a0, false, b0, (short)0, acc[0], false, false);
        acc[1] = __builtin_amdgcn_wmma_f32_16x16x32_bf16(false, a0, false, b1, (short)0, acc[1], false, false);
        acc[2] = __builtin_amdgcn_wmma_f32_16x16x32_bf16(false, a1, false, b0, (short)0, acc[2], false, false);
        acc[3] = __builtin_amdgcn_wmma_f32_16x16x32_bf16(false, a1, false, b1, (short)0, acc[3], false, false);
    }
    int mh = (lane >> 4) * 8;
#pragma unroll
    for (int i = 0; i < 2; i++) {
#pragma unroll
        for (int j = 0; j < 2; j++) {
#pragma unroll
            for (int v = 0; v < 8; v++) {
                int row = mt * 32 + i * 16 + v + mh;
                int col = nt * 32 + j * 16 + mi;
                ip[((size_t)bb * NEDGES + row) * NEDGES + col] = acc[i * 2 + j][v];
            }
        }
    }
}

// ---------------------------------------------------------------- wave row reductions (dot / sq / sum)
#define RD_DOT 0
#define RD_SQ  1
#define RD_SUM 2
__global__ __launch_bounds__(256) void rowreduce_kernel(const float* __restrict__ src,
                                                        const float* __restrict__ vec,
                                                        float* __restrict__ dst, int mode) {
    int w    = (blockIdx.x * blockDim.x + threadIdx.x) >> 5;   // grid sized exactly rows/8
    int lane = threadIdx.x & 31;
    const float* row = src + (size_t)w * C_;
    float s = 0.0f;
    if (mode == RD_DOT)     { for (int c = lane; c < C_; c += 32) s += row[c] * vec[c]; }
    else if (mode == RD_SQ) { for (int c = lane; c < C_; c += 32) s += row[c] * row[c]; }
    else                    { for (int c = lane; c < C_; c += 32) s += row[c]; }
    for (int o = 16; o; o >>= 1) s += __shfl_xor(s, o, 32);
    if (!lane) dst[w] = s;
}

__global__ void combine16_kernel(const float* __restrict__ src, float* __restrict__ dst, int rows) {
    int r = blockIdx.x * blockDim.x + threadIdx.x;
    if (r < rows) {
        float s = 0.0f;
        for (int b = 0; b < B_; b++) s += src[(size_t)r * B_ + b];
        dst[r] = s;
    }
}

// ---------------------------------------------------------------- attention: alpha + segment softmax
__global__ void alpha_kernel(const float* __restrict__ s_node, const float* __restrict__ s_edge,
                             const int* __restrict__ node_idx, const int* __restrict__ edge_idx,
                             float* __restrict__ alpha, float* __restrict__ amax,
                             int* __restrict__ Dcnt) {
    int idx = blockIdx.x * blockDim.x + threadIdx.x;   // exact 65536
    int e = idx >> 4, b = idx & 15;
    int n = node_idx[e], m = edge_idx[e];
    float v = s_node[n * B_ + b] + s_edge[m * B_ + b];
    v = v > 0.0f ? v : NEG_SLOPE * v;
    alpha[idx] = v;
    atomicMaxFloat(&amax[n * B_ + b], v);
    if (b == 0) atomicAdd(&Dcnt[n], 1);
}

__global__ void dconv_kernel(const int* __restrict__ Dcnt, float* __restrict__ Df) {
    int n = blockIdx.x * blockDim.x + threadIdx.x;     // exact 1024
    Df[n] = (float)Dcnt[n];
}

__global__ void exp_kernel(float* __restrict__ alpha, const float* __restrict__ amax,
                           const int* __restrict__ node_idx, float* __restrict__ denom) {
    int idx = blockIdx.x * blockDim.x + threadIdx.x;
    int e = idx >> 4, b = idx & 15;
    int n = node_idx[e];
    float ex = __expf(alpha[idx] - amax[n * B_ + b]);
    alpha[idx] = ex;
    atomicAdd(&denom[n * B_ + b], ex);
}

__global__ void norm_kernel(float* __restrict__ alpha, const float* __restrict__ denom,
                            const int* __restrict__ node_idx) {
    int idx = blockIdx.x * blockDim.x + threadIdx.x;
    int e = idx >> 4, b = idx & 15;
    int n = node_idx[e];
    alpha[idx] = alpha[idx] / (denom[n * B_ + b] + EPS_SM);
}

// ---------------------------------------------------------------- pass 1: nodes -> hyperedges
__global__ __launch_bounds__(256) void pass1_kernel(const float* __restrict__ x1,
                                                    const float* __restrict__ alpha,
                                                    const int* __restrict__ node_idx,
                                                    const int* __restrict__ edge_idx,
                                                    const float* __restrict__ Bn,
                                                    float* __restrict__ out_e) {
    __shared__ int snode[NINC];
    int m  = blockIdx.x;
    int lo = lower_bound_i(edge_idx, NINC, m);
    int hi = lower_bound_i(edge_idx, NINC, m + 1);
    int cnt = hi - lo;
    for (int i = threadIdx.x; i < cnt; i += blockDim.x) snode[i] = node_idx[lo + i];
    __syncthreads();
    float bnv = Bn[m];
    int t = threadIdx.x;
    for (int b = 0; b < B_; b++) {
        float a0 = 0.0f, a1 = 0.0f;
        for (int i = 0; i < cnt; i++) {
            float al = alpha[(size_t)(lo + i) * B_ + b];
            const float* row = x1 + ((size_t)snode[i] * B_ + b) * C_;
            a0 += al * row[2 * t]; a1 += al * row[2 * t + 1];
        }
        size_t o = ((size_t)m * B_ + b) * C_ + 2 * t;
        out_e[o] = bnv * a0; out_e[o + 1] = bnv * a1;
    }
}

// ---------------------------------------------------------------- pass 2: hyperedges -> nodes (into d_out [B][N][C])
__global__ __launch_bounds__(256) void pass2_kernel(const float* __restrict__ out_e,
                                                    const float* __restrict__ alpha,
                                                    const int* __restrict__ node_idx,
                                                    const int* __restrict__ edge_idx,
                                                    const float* __restrict__ Df,
                                                    float* __restrict__ out) {
    int e = blockIdx.x;                                // 4096 incidences
    int n = node_idx[e], m = edge_idx[e];
    float dn = Df[n];
    int t = threadIdx.x;
    for (int b = 0; b < B_; b++) {
        float al = alpha[(size_t)e * B_ + b] * dn;
        const float* src = out_e + ((size_t)m * B_ + b) * C_;
        float* dst = out + ((size_t)b * NNODES + n) * C_;
        atomicAdd(dst + 2 * t,     al * src[2 * t]);
        atomicAdd(dst + 2 * t + 1, al * src[2 * t + 1]);
    }
}

// ---------------------------------------------------------------- losses
__global__ void diff_kernel(const float* __restrict__ tn, const float* __restrict__ te,
                            const int* __restrict__ node_idx, const int* __restrict__ edge_idx,
                            float* __restrict__ scal) {
    __shared__ float red[256];
    float s = 0.0f;
    for (int e = threadIdx.x; e < NINC; e += 256)
        s += tn[node_idx[e]] - te[edge_idx[e]];
    red[threadIdx.x] = s; __syncthreads();
    for (int o = 128; o; o >>= 1) {
        if (threadIdx.x < o) red[threadIdx.x] += red[threadIdx.x + o];
        __syncthreads();
    }
    if (threadIdx.x == 0) scal[0] = red[0];
}

__global__ void hyper_kernel(const float* __restrict__ ip, const float* __restrict__ sq,
                             float* __restrict__ hpart) {
    __shared__ float red[256];
    int idx = blockIdx.x * 256 + threadIdx.x;          // 65536 (k,m) pairs
    int k = idx >> 8, m = idx & 255;
    float s = 0.0f;
    for (int b = 0; b < B_; b++) {
        float ipv = ip[((size_t)b * NEDGES + k) * NEDGES + m];
        float sk = sq[k * B_ + b], sm = sq[m * B_ + b];
        float nk = sqrtf(fmaxf(sk, 1e-12f)), nm = sqrtf(fmaxf(sm, 1e-12f));
        float ap = ipv / fmaxf(nk * nm, 1e-12f);
        float dist = sqrtf(fmaxf(sk + sm - 2.0f * ipv, 1e-12f));
        s += ap * dist + (1.0f - ap) * fmaxf(MARGIN - dist, 0.0f);
    }
    s = fabsf(s * (1.0f / (float)B_));
    red[threadIdx.x] = s; __syncthreads();
    for (int o = 128; o; o >>= 1) {
        if (threadIdx.x < o) red[threadIdx.x] += red[threadIdx.x + o];
        __syncthreads();
    }
    if (threadIdx.x == 0) hpart[blockIdx.x] = red[0];
}

__global__ void final_kernel(const float* __restrict__ scal, const float* __restrict__ hpart,
                             float* __restrict__ out) {
    __shared__ float red[256];
    red[threadIdx.x] = hpart[threadIdx.x]; __syncthreads();
    for (int o = 128; o; o >>= 1) {
        if (threadIdx.x < o) red[threadIdx.x] += red[threadIdx.x + o];
        __syncthreads();
    }
    if (threadIdx.x == 0) {
        float loss_hyper = red[0] / ((float)(NEDGES + 1) * (float)(NEDGES + 1));
        float mean_diff  = scal[0] / (float)((size_t)NINC * B_ * C_);
        out[OUT_ELEMS] = fabsf(mean_diff) + loss_hyper;
    }
}

// ---------------------------------------------------------------- launch
extern "C" void kernel_launch(void* const* d_in, const int* in_sizes, int n_in,
                              void* d_out, int out_size, void* d_ws, size_t ws_size,
                              hipStream_t stream) {
    const float* x      = (const float*)d_in[0];
    const float* weight = (const float*)d_in[1];
    const float* att    = (const float*)d_in[2];     // 1024 floats
    const int*   nidx   = (const int*)d_in[3];
    const int*   eidx   = (const int*)d_in[4];
    float* out = (float*)d_out;

    char* ws = (char*)d_ws;
    unsigned short* xbf  = (unsigned short*)(ws + 0);            // 16 MB
    unsigned short* wt   = (unsigned short*)(ws + 16777216);     // 0.5 MB
    float* x1            = (float*)(ws + 17301504);              // 32 MB  [N][B][C]
    float* esum          = (float*)(ws + 50855936);              // 8 MB   [M][B][C]
    unsigned short* esb  = (unsigned short*)(ws + 59244544);     // 4 MB   [B][M][C] bf16
    float* ip            = (float*)(ws + 63438848);              // 4 MB   [B][M][M]
    float* out_e         = (float*)(ws + 67633152);              // 8 MB   [M][B][C]
    float* alpha         = (float*)(ws + 76021760);              // 256 KB [E][B]
    float* amax          = (float*)(ws + 76283904);              // 64 KB  [N][B]
    float* denom         = (float*)(ws + 76349440);              // 64 KB
    float* s_node        = (float*)(ws + 76414976);              // 64 KB
    float* s_edge        = (float*)(ws + 76480512);              // 16 KB
    float* sq            = (float*)(ws + 76496896);              // 16 KB  [M][B]
    float* Bn            = (float*)(ws + 76513280);              // 1 KB
    int*   Dcnt          = (int*)  (ws + 76514304);              // 4 KB
    float* Df            = (float*)(ws + 76518400);              // 4 KB
    float* rsn           = (float*)(ws + 76522496);              // 64 KB  rowsums x1
    float* rse           = (float*)(ws + 76588032);              // 16 KB  rowsums esum
    float* tn            = (float*)(ws + 76604416);              // 4 KB
    float* te            = (float*)(ws + 76608512);              // 1 KB
    float* scal          = (float*)(ws + 76609536);
    float* hpart         = (float*)(ws + 76609792);              // 1 KB

    // init
    fill_u32_kernel<<<2048, 256, 0, stream>>>((unsigned int*)out, (size_t)OUT_ELEMS + 1, 0u);
    fill_u32_kernel<<<64, 256, 0, stream>>>((unsigned int*)denom, NNODES * B_, 0u);
    fill_u32_kernel<<<64, 256, 0, stream>>>((unsigned int*)amax, NNODES * B_, 0xFF800000u); // -inf
    fill_u32_kernel<<<4, 256, 0, stream>>>((unsigned int*)Dcnt, NNODES, 0u);

    // bf16 staging
    convert_x_kernel<<<4096, 256, 0, stream>>>(x, xbf);
    convert_wt_kernel<<<1024, 256, 0, stream>>>(weight, wt);

    // GEMM 1 (WMMA bf16, 32x64 register-blocked): x1[n][b][c]
    gemm1_kernel<<<512, 256, 0, stream>>>(xbf, wt, x1);

    // edge segment sums, Bn, bf16 copy
    edge_sums_kernel<<<NEDGES, 256, 0, stream>>>(x1, nidx, eidx, esum, esb, Bn);

    // GEMM 2 (WMMA bf16, 32x32 register-blocked): ip[b] = G_b G_b^T
    gemm2_kernel<<<128, 256, 0, stream>>>(esb, ip);

    // row reductions
    rowreduce_kernel<<<2048, 256, 0, stream>>>(x1,  att,        s_node, RD_DOT);  // 16384 rows
    rowreduce_kernel<<<512,  256, 0, stream>>>(esum, att + C_,  s_edge, RD_DOT);  // 4096 rows
    rowreduce_kernel<<<512,  256, 0, stream>>>(esum, att,       sq,     RD_SQ);
    rowreduce_kernel<<<2048, 256, 0, stream>>>(x1,  att,        rsn,    RD_SUM);
    rowreduce_kernel<<<512,  256, 0, stream>>>(esum, att,       rse,    RD_SUM);
    combine16_kernel<<<4, 256, 0, stream>>>(rsn, tn, NNODES);
    combine16_kernel<<<1, 256, 0, stream>>>(rse, te, NEDGES);

    // attention + segment softmax over nodes
    alpha_kernel<<<256, 256, 0, stream>>>(s_node, s_edge, nidx, eidx, alpha, amax, Dcnt);
    dconv_kernel<<<4, 256, 0, stream>>>(Dcnt, Df);
    exp_kernel<<<256, 256, 0, stream>>>(alpha, amax, nidx, denom);
    norm_kernel<<<256, 256, 0, stream>>>(alpha, denom, nidx);

    // message passing
    pass1_kernel<<<NEDGES, 256, 0, stream>>>(x1, alpha, nidx, eidx, Bn, out_e);
    pass2_kernel<<<NINC, 256, 0, stream>>>(out_e, alpha, nidx, eidx, Df, out);

    // losses
    diff_kernel<<<1, 256, 0, stream>>>(tn, te, nidx, eidx, scal);
    hyper_kernel<<<256, 256, 0, stream>>>(ip, sq, hpart);
    final_kernel<<<1, 256, 0, stream>>>(scal, hpart, out);
}
